// CrossAttention_87376814670423
// MI455X (gfx1250) — matrix operand order repaired and checked by reading, not compiled, not run on hardware
//
#include <hip/hip_runtime.h>
#include <hip/hip_bf16.h>

// ---------------------------------------------------------------------------
// Types
// ---------------------------------------------------------------------------
typedef __bf16 bf16_t;
typedef __attribute__((ext_vector_type(16))) __bf16 v16bf;
typedef __attribute__((ext_vector_type(8)))  __bf16 v8bf;
typedef __attribute__((ext_vector_type(4)))  __bf16 v4bf;
typedef __attribute__((ext_vector_type(8)))  float  v8f;
typedef __attribute__((ext_vector_type(4)))  int    v4i;

// Problem constants (from the reference)
#define BATCH   8
#define NQ      4096
#define QD      1024
#define CD      768
#define IDIM    1024
#define NHEADS  16
#define DHEAD   64
#define CTX     81
#define ORG     77
#define RANK    32
#define ATT_SCALE 0.125f   // 64^-0.5

// ---------------------------------------------------------------------------
// gfx1250 async global->LDS path (ASYNCcnt), guarded so the build can't break
// Builtin signature (from hipcc diagnostic): b128 takes typed v4i pointers,
// (global/AS1 src, LDS/AS3 dst, imm offset, imm cpol).
// ---------------------------------------------------------------------------
#if defined(__gfx1250__) &&                                            \
    __has_builtin(__builtin_amdgcn_global_load_async_to_lds_b128) &&   \
    __has_builtin(__builtin_amdgcn_s_wait_asynccnt)
#define HAVE_ASYNC_LDS 1
#else
#define HAVE_ASYNC_LDS 0
#endif

#if HAVE_ASYNC_LDS
#define GLOBAL_AS __attribute__((address_space(1)))
#define LDS_AS    __attribute__((address_space(3)))
__device__ __forceinline__ void async_ld16(void* lds, const void* g) {
  __builtin_amdgcn_global_load_async_to_lds_b128(
      (GLOBAL_AS v4i*)g, (LDS_AS v4i*)lds, 0, 0);
}
#define ASYNC_WAIT(n) __builtin_amdgcn_s_wait_asynccnt(n)
#else
#define ASYNC_WAIT(n)
#endif

// ---------------------------------------------------------------------------
// WMMA fragment helpers (this file's contiguous k order, same for A and B; not the ISA table order)
//   A 16x32 bf16: lane m=lane&15; elems j -> K = 16*half + j (contiguous 16)
//   B 32x16 bf16: lane n=lane&15; elems j -> K = 16*half + j (contiguous 16)
// ---------------------------------------------------------------------------
__device__ __forceinline__ v16bf load_a_frag(const bf16_t* base, int stride) {
  int lane = threadIdx.x & 31;
  int m = lane & 15, hf = lane >> 4;
  const bf16_t* p = base + (size_t)m * stride;
  return *(const v16bf*)(p + 16 * hf);
}

__device__ __forceinline__ v16bf load_b_frag(const bf16_t* base, int stride) {
  int lane = threadIdx.x & 31;
  int n = lane & 15, hf = lane >> 4;
  return *(const v16bf*)(base + (size_t)n * stride + 16 * hf);
}

__device__ __forceinline__ v8f wmma_bf16(v16bf a, v16bf b, v8f c) {
  return __builtin_amdgcn_wmma_f32_16x16x32_bf16(false, a, false, b, (short)0, c,
                                                 false, false);
}

// ---------------------------------------------------------------------------
// Kernel 1: fold LoRA into weight, cast to bf16, store TRANSPOSED [N][K].
//   out[c][r] = W[r][c] + (alpha/RANK) * sum_t down[r][t]*up[t][c]
// Transposed storage makes GEMM B-tiles row-contiguous in K so both A and B
// staging are pure 16-byte async copies (no scatter/transpose in LDS).
// ---------------------------------------------------------------------------
__global__ __launch_bounds__(256) void fold_weight_kernel(
    const float* __restrict__ W, const float* __restrict__ down,
    const float* __restrict__ up, const float* __restrict__ alpha,
    bf16_t* __restrict__ outT, int R, int C, int useLora) {
  int idx = blockIdx.x * 256 + threadIdx.x;
  if (idx >= R * C) return;
  int r = idx / C, c = idx % C;
  float v = W[idx];
  if (useLora) {
    float s = alpha[0] * (1.0f / (float)RANK);  // LORA_W == 1
    float acc = 0.0f;
#pragma unroll
    for (int t = 0; t < RANK; ++t) acc += down[r * RANK + t] * up[t * C + c];
    v += acc * s;
  }
  outT[(size_t)c * R + r] = (bf16_t)v;
}

// ---------------------------------------------------------------------------
// Kernel 2: f32 -> bf16 cast
// ---------------------------------------------------------------------------
__global__ __launch_bounds__(256) void cast_bf16_kernel(
    const float* __restrict__ in, bf16_t* __restrict__ out, long n) {
  long base = ((long)blockIdx.x * 256 + threadIdx.x) * 4;
#pragma unroll
  for (int j = 0; j < 4; ++j)
    if (base + j < n) out[base + j] = (bf16_t)in[base + j];
}

// ---------------------------------------------------------------------------
// Kernel 3: bf16 WMMA GEMM  C[M,N] = A[M,K] * Bt[N,K]^T (+bias)
//   256 threads (8 waves), block tile 128x128, K-step 32, double-buffered
//   LDS fed by async global->LDS copies (ASYNCcnt ping-pong pipeline).
//   Wave (wr,wc) owns a 64x32 sub-tile: 4 M-tiles x 2 N-tiles = 8 WMMAs/step.
// ---------------------------------------------------------------------------
template <bool OUT_F32>
__global__ __launch_bounds__(256) void gemm_bf16_kernel(
    const bf16_t* __restrict__ A, const bf16_t* __restrict__ Bt,
    void* __restrict__ Cout, const float* __restrict__ bias,
    int M, int N, int K) {
  __shared__ __align__(32) bf16_t As[2][128][32];  // [buf][m][k]
  __shared__ __align__(32) bf16_t Bs[2][128][32];  // [buf][n][k]

  const int tid = threadIdx.x, lane = tid & 31, wv = tid >> 5;
  const int m0 = blockIdx.y * 128, n0 = blockIdx.x * 128;
  const int wr = wv >> 2, wc = wv & 3;

  // staging coords: thread -> (row 0..127, 16-elem chunk)
  const int sr = tid >> 1;
  const int sc = (tid & 1) * 16;
  int arow = m0 + sr;
  if (arow >= M) arow = M - 1;  // clamp: garbage rows are never stored
  const bf16_t* gA = A + (size_t)arow * K + sc;
  const bf16_t* gB = Bt + (size_t)(n0 + sr) * K + sc;

  v8f acc[4][2] = {};

  auto stage = [&](int ks, int p) {
    const bf16_t* pa = gA + ks * 32;
    const bf16_t* pb = gB + ks * 32;
    bf16_t* la = &As[p][sr][sc];
    bf16_t* lb = &Bs[p][sr][sc];
#if HAVE_ASYNC_LDS
    async_ld16(la, pa);
    async_ld16(la + 8, pa + 8);
    async_ld16(lb, pb);
    async_ld16(lb + 8, pb + 8);
#else
    *(v8bf*)la = *(const v8bf*)pa;
    *(v8bf*)(la + 8) = *(const v8bf*)(pa + 8);
    *(v8bf*)lb = *(const v8bf*)pb;
    *(v8bf*)(lb + 8) = *(const v8bf*)(pb + 8);
#endif
  };

  auto compute = [&](int p) {
    v16bf b0 = load_b_frag(&Bs[p][wc * 32][0], 32);
    v16bf b1 = load_b_frag(&Bs[p][wc * 32 + 16][0], 32);
#pragma unroll
    for (int mt = 0; mt < 4; ++mt) {
      v16bf af = load_a_frag(&As[p][wr * 64 + mt * 16][0], 32);
      acc[mt][0] = wmma_bf16(af, b0, acc[mt][0]);
      acc[mt][1] = wmma_bf16(af, b1, acc[mt][1]);
    }
  };

  const int KS = K >> 5;
  stage(0, 0);
  for (int ks = 0; ks < KS - 1; ++ks) {
    stage(ks + 1, (ks + 1) & 1);
    ASYNC_WAIT(4);       // previous stage's 4 copies retired (in-order)
    __syncthreads();
    compute(ks & 1);
    __syncthreads();     // buffer (ks&1) free for stage ks+2
  }
  ASYNC_WAIT(0);
  __syncthreads();
  compute((KS - 1) & 1);

  // Epilogue. C layout: elem r -> row M = r + 8*half, col N = lane&15.
  const int hf = lane >> 4, nn = lane & 15;
#pragma unroll
  for (int nt = 0; nt < 2; ++nt) {
    int col = n0 + wc * 32 + nt * 16 + nn;
    float badd = (OUT_F32 && bias != nullptr) ? bias[col] : 0.0f;
#pragma unroll
    for (int mt = 0; mt < 4; ++mt) {
#pragma unroll
      for (int r = 0; r < 8; ++r) {
        int row = m0 + wr * 64 + mt * 16 + 8 * hf + r;
        if (row < M) {
          if (OUT_F32)
            ((float*)Cout)[(size_t)row * N + col] = acc[mt][nt][r] + badd;
          else
            ((bf16_t*)Cout)[(size_t)row * N + col] = (bf16_t)acc[mt][nt][r];
        }
      }
    }
  }
}

// ---------------------------------------------------------------------------
// Kernel 4: fused dual-softmax attention.
//   Grid: (NQ/64, NHEADS, BATCH), 128 threads (4 waves, 16 queries each).
//   Keys 0..76 = org context, 80..83 = IPA keys, rest zero-padded.
//   S = Q K^T * scale; softmax(org) and softmax(ipa) separately; P is
//   pre-divided by its row sum so a single P(16x96) x V(96x64) product
//   yields attn_org + attn_ipa.
// ---------------------------------------------------------------------------
__global__ __launch_bounds__(128) void attn_kernel(
    const bf16_t* __restrict__ q, const bf16_t* __restrict__ k_org,
    const bf16_t* __restrict__ v_org, const bf16_t* __restrict__ k_ipa,
    const bf16_t* __restrict__ v_ipa, bf16_t* __restrict__ out) {
  __shared__ __align__(32) bf16_t Kl[96][64];      // [key][d]
  __shared__ __align__(32) bf16_t Vt[64][96];      // [d][key] (transposed)
  __shared__ __align__(32) bf16_t Pl[4][16][96];   // per-wave P

  int tid = threadIdx.x;
  int b = blockIdx.z, hd = blockIdx.y, qb = blockIdx.x;

  // --- stage K / V^T into LDS (96 keys x 64 dims, zero padded) ---
  for (int idx = tid; idx < 96 * 16; idx += 128) {
    int j = idx >> 4;             // key 0..95
    int d = (idx & 15) << 2;      // dim 0..60 step 4
    v4bf kv = {}, vv = {};
    if (j < ORG) {
      size_t o = ((size_t)(b * CTX + j)) * IDIM + hd * DHEAD + d;
      kv = *(const v4bf*)(k_org + o);
      vv = *(const v4bf*)(v_org + o);
    } else if (j >= 80 && j < 84) {
      size_t o = ((size_t)(b * CTX + ORG + (j - 80))) * IDIM + hd * DHEAD + d;
      kv = *(const v4bf*)(k_ipa + o);
      vv = *(const v4bf*)(v_ipa + o);
    }
    *(v4bf*)&Kl[j][d] = kv;
#pragma unroll
    for (int i = 0; i < 4; ++i) Vt[d + i][j] = vv[i];
  }
  __syncthreads();

  int lane = tid & 31, wv = tid >> 5;
  int hf = lane >> 4, m16 = lane & 15;

  const bf16_t* qbase =
      q + ((size_t)(b * NQ + qb * 64 + wv * 16)) * IDIM + hd * DHEAD;

  // --- S = Q K^T : 6 key tiles x 2 d-chunks ---
  v8f s[6] = {};
#pragma unroll
  for (int kc = 0; kc < 2; ++kc) {
    v16bf aq = load_a_frag(qbase + kc * 32, IDIM);
#pragma unroll
    for (int t = 0; t < 6; ++t) {
      v16bf bk = load_b_frag(&Kl[t * 16][kc * 32], 64);
      s[t] = wmma_bf16(aq, bk, s[t]);
    }
  }

  // --- scale + mask into registers ---
  float sv[6][8];
#pragma unroll
  for (int t = 0; t < 6; ++t) {
    int col = t * 16 + m16;  // global key for t<5; (80 + m16) for t==5
    bool valid = (t < 5) ? (col < ORG) : (m16 < 4);
#pragma unroll
    for (int r = 0; r < 8; ++r)
      sv[t][r] = valid ? s[t][r] * ATT_SCALE : -3.0e38f;
  }

  // --- softmax over org keys (tiles 0..4) ---
  float mx1[8], sum1[8];
#pragma unroll
  for (int r = 0; r < 8; ++r) {
    float m = sv[0][r];
#pragma unroll
    for (int t = 1; t < 5; ++t) m = fmaxf(m, sv[t][r]);
    mx1[r] = m;
  }
#pragma unroll
  for (int off = 1; off < 16; off <<= 1)
#pragma unroll
    for (int r = 0; r < 8; ++r)
      mx1[r] = fmaxf(mx1[r], __shfl_xor(mx1[r], off, 32));
#pragma unroll
  for (int r = 0; r < 8; ++r) sum1[r] = 0.0f;
#pragma unroll
  for (int t = 0; t < 5; ++t)
#pragma unroll
    for (int r = 0; r < 8; ++r) {
      float e = __expf(sv[t][r] - mx1[r]);
      sv[t][r] = e;
      sum1[r] += e;
    }
#pragma unroll
  for (int off = 1; off < 16; off <<= 1)
#pragma unroll
    for (int r = 0; r < 8; ++r) sum1[r] += __shfl_xor(sum1[r], off, 32);

  // --- softmax over IPA keys (tile 5) ---
  float mx2[8], sum2[8];
#pragma unroll
  for (int r = 0; r < 8; ++r) mx2[r] = sv[5][r];
#pragma unroll
  for (int off = 1; off < 16; off <<= 1)
#pragma unroll
    for (int r = 0; r < 8; ++r)
      mx2[r] = fmaxf(mx2[r], __shfl_xor(mx2[r], off, 32));
#pragma unroll
  for (int r = 0; r < 8; ++r) {
    float e = __expf(sv[5][r] - mx2[r]);
    sv[5][r] = e;
    sum2[r] = e;
  }
#pragma unroll
  for (int off = 1; off < 16; off <<= 1)
#pragma unroll
    for (int r = 0; r < 8; ++r) sum2[r] += __shfl_xor(sum2[r], off, 32);

  // --- write pre-normalized P to LDS (C layout -> row = r + 8*half) ---
#pragma unroll
  for (int t = 0; t < 5; ++t)
#pragma unroll
    for (int r = 0; r < 8; ++r)
      Pl[wv][8 * hf + r][t * 16 + m16] = (bf16_t)(sv[t][r] / sum1[r]);
#pragma unroll
  for (int r = 0; r < 8; ++r)
    Pl[wv][8 * hf + r][80 + m16] = (bf16_t)(sv[5][r] / sum2[r]);

  // --- O = P (16x96) x V (96x64) ---
  v8f o[4] = {};
#pragma unroll
  for (int kc = 0; kc < 3; ++kc) {
    v16bf ap = load_a_frag(&Pl[wv][0][kc * 32], 96);
#pragma unroll
    for (int nt = 0; nt < 4; ++nt) {
      v16bf bv = load_b_frag(&Vt[nt * 16][kc * 32], 96);
      o[nt] = wmma_bf16(ap, bv, o[nt]);
    }
  }

  // --- store ---
  bf16_t* ob = out + ((size_t)(b * NQ + qb * 64 + wv * 16 + 8 * hf)) * IDIM +
               hd * DHEAD;
#pragma unroll
  for (int nt = 0; nt < 4; ++nt)
#pragma unroll
    for (int r = 0; r < 8; ++r)
      ob[(size_t)r * IDIM + nt * 16 + m16] = (bf16_t)o[nt][r];
}

// ---------------------------------------------------------------------------
// Host-side orchestration
// ---------------------------------------------------------------------------
extern "C" void kernel_launch(void* const* d_in, const int* in_sizes, int n_in,
                              void* d_out, int out_size, void* d_ws,
                              size_t ws_size, hipStream_t stream) {
  (void)in_sizes; (void)n_in; (void)out_size; (void)ws_size;
  const float* x       = (const float*)d_in[0];
  const float* context = (const float*)d_in[1];
  const float* Wq      = (const float*)d_in[2];
  const float* Wk      = (const float*)d_in[3];
  const float* Wv      = (const float*)d_in[4];
  const float* Wo      = (const float*)d_in[5];
  const float* bo      = (const float*)d_in[6];
  const float* q_down  = (const float*)d_in[7];
  const float* q_up    = (const float*)d_in[8];
  const float* q_alpha = (const float*)d_in[9];
  const float* k_down  = (const float*)d_in[10];
  const float* k_up    = (const float*)d_in[11];
  const float* k_alpha = (const float*)d_in[12];
  const float* v_down  = (const float*)d_in[13];
  const float* v_up    = (const float*)d_in[14];
  const float* v_alpha = (const float*)d_in[15];
  const float* o_down  = (const float*)d_in[16];
  const float* o_up    = (const float*)d_in[17];
  const float* o_alpha = (const float*)d_in[18];
  const float* Wk_ipa  = (const float*)d_in[19];
  const float* Wv_ipa  = (const float*)d_in[20];

  char* ws = (char*)d_ws;
  size_t off = 0;
  auto alloc = [&](size_t bytes) -> void* {
    off = (off + 255) & ~(size_t)255;
    void* p = ws + off;
    off += bytes;
    return p;
  };

  bf16_t* Wq_e  = (bf16_t*)alloc((size_t)QD * IDIM * 2);   // transposed [N][K]
  bf16_t* Wk_e  = (bf16_t*)alloc((size_t)CD * IDIM * 2);
  bf16_t* Wv_e  = (bf16_t*)alloc((size_t)CD * IDIM * 2);
  bf16_t* Wo_e  = (bf16_t*)alloc((size_t)IDIM * QD * 2);
  bf16_t* Wki_e = (bf16_t*)alloc((size_t)CD * IDIM * 2);
  bf16_t* Wvi_e = (bf16_t*)alloc((size_t)CD * IDIM * 2);
  bf16_t* x_bf  = (bf16_t*)alloc((size_t)BATCH * NQ * QD * 2);
  bf16_t* c_bf  = (bf16_t*)alloc((size_t)BATCH * CTX * CD * 2);
  bf16_t* q_bf  = (bf16_t*)alloc((size_t)BATCH * NQ * IDIM * 2);
  bf16_t* k_org = (bf16_t*)alloc((size_t)BATCH * CTX * IDIM * 2);
  bf16_t* v_org = (bf16_t*)alloc((size_t)BATCH * CTX * IDIM * 2);
  bf16_t* k_ipa = (bf16_t*)alloc((size_t)BATCH * CTX * IDIM * 2);
  bf16_t* v_ipa = (bf16_t*)alloc((size_t)BATCH * CTX * IDIM * 2);
  bf16_t* a_out = (bf16_t*)alloc((size_t)BATCH * NQ * IDIM * 2);

  // 1) fold LoRA into weights (bf16, transposed)
  fold_weight_kernel<<<(QD * IDIM + 255) / 256, 256, 0, stream>>>(
      Wq, q_down, q_up, q_alpha, Wq_e, QD, IDIM, 1);
  fold_weight_kernel<<<(CD * IDIM + 255) / 256, 256, 0, stream>>>(
      Wk, k_down, k_up, k_alpha, Wk_e, CD, IDIM, 1);
  fold_weight_kernel<<<(CD * IDIM + 255) / 256, 256, 0, stream>>>(
      Wv, v_down, v_up, v_alpha, Wv_e, CD, IDIM, 1);
  fold_weight_kernel<<<(IDIM * QD + 255) / 256, 256, 0, stream>>>(
      Wo, o_down, o_up, o_alpha, Wo_e, IDIM, QD, 1);
  fold_weight_kernel<<<(CD * IDIM + 255) / 256, 256, 0, stream>>>(
      Wk_ipa, q_down, q_up, q_alpha, Wki_e, CD, IDIM, 0);
  fold_weight_kernel<<<(CD * IDIM + 255) / 256, 256, 0, stream>>>(
      Wv_ipa, q_down, q_up, q_alpha, Wvi_e, CD, IDIM, 0);

  // 2) cast activations
  {
    long n = (long)BATCH * NQ * QD;
    cast_bf16_kernel<<<(int)((n + 1023) / 1024), 256, 0, stream>>>(x, x_bf, n);
  }
  {
    long n = (long)BATCH * CTX * CD;
    cast_bf16_kernel<<<(int)((n + 1023) / 1024), 256, 0, stream>>>(context, c_bf, n);
  }

  // 3) Q projection: [32768,1024] x [1024,1024]
  {
    dim3 grid(IDIM / 128, (BATCH * NQ) / 128);
    gemm_bf16_kernel<false><<<grid, 256, 0, stream>>>(
        x_bf, Wq_e, (void*)q_bf, nullptr, BATCH * NQ, IDIM, QD);
  }

  // 4) K/V projections over all 81 context rows: [648,768] x [768,1024]
  {
    int Mkv = BATCH * CTX;
    dim3 grid(IDIM / 128, (Mkv + 127) / 128);
    gemm_bf16_kernel<false><<<grid, 256, 0, stream>>>(
        c_bf, Wk_e, (void*)k_org, nullptr, Mkv, IDIM, CD);
    gemm_bf16_kernel<false><<<grid, 256, 0, stream>>>(
        c_bf, Wv_e, (void*)v_org, nullptr, Mkv, IDIM, CD);
    gemm_bf16_kernel<false><<<grid, 256, 0, stream>>>(
        c_bf, Wki_e, (void*)k_ipa, nullptr, Mkv, IDIM, CD);
    gemm_bf16_kernel<false><<<grid, 256, 0, stream>>>(
        c_bf, Wvi_e, (void*)v_ipa, nullptr, Mkv, IDIM, CD);
  }

  // 5) attention (org + IPA fused)
  {
    dim3 grid(NQ / 64, NHEADS, BATCH);
    attn_kernel<<<grid, 128, 0, stream>>>(q_bf, k_org, v_org, k_ipa, v_ipa,
                                          a_out);
  }

  // 6) output projection + bias -> f32 d_out
  {
    dim3 grid(QD / 128, (BATCH * NQ) / 128);
    gemm_bf16_kernel<true><<<grid, 256, 0, stream>>>(
        a_out, Wo_e, d_out, bo, BATCH * NQ, QD, IDIM);
  }
}
